// PatchSelect_33560874451584
// MI455X (gfx1250) — matrix operand order repaired and checked by reading, not compiled, not run on hardware
//
#include <hip/hip_runtime.h>
#include <hip/hip_bf16.h>
#include <math.h>

typedef __bf16 bf16_t;
typedef __attribute__((ext_vector_type(16))) __bf16 v16bf;
typedef __attribute__((ext_vector_type(8)))  float  v8f;

#define DIMD 1024
#define DIMH 512
#define DIMB 16
#define DIMN 4096
#define DIMO 4096
#define KSEL 200

union ABfrag { v16bf v; uint4 q[2]; };

// ---------------------------------------------------------------------------
// fp32 -> bf16 conversion (grid-stride)
// ---------------------------------------------------------------------------
__global__ __launch_bounds__(256) void cvt_bf16_kernel(const float* __restrict__ src,
                                                       bf16_t* __restrict__ dst, int n) {
    for (int i = blockIdx.x * 256 + threadIdx.x; i < n; i += gridDim.x * 256)
        dst[i] = (bf16_t)src[i];
}

// ---------------------------------------------------------------------------
// Fused score kernel: scores[b,n] = W2 . gelu(W1 x + b1) + b2
// One block (8 waves) per 64-token tile: 4 M-tiles of 16, W1 B-fragment
// reused across 4 WMMAs; K=1024 via v_wmma_f32_16x16x32_bf16.
// ---------------------------------------------------------------------------
__global__ __launch_bounds__(256) void score_kernel(const float* __restrict__ x,
                                                    const bf16_t* __restrict__ W1b,
                                                    const float* __restrict__ b1,
                                                    const float* __restrict__ W2,
                                                    const float* __restrict__ b2,
                                                    float* __restrict__ scores) {
    __shared__ bf16_t Atile[64][DIMD + 8];   // 129 KB, padded rows -> bank spread
    __shared__ float  spart[8][64];

    const long rowbase = (long)blockIdx.x * 64;
    const int  tid     = threadIdx.x;

    // stage x tile (64 x 1024 fp32) as bf16 into LDS
    for (int i = tid; i < 64 * DIMD / 4; i += 256) {
        const int r = (i * 4) / DIMD;
        const int c = (i * 4) % DIMD;
        const float4 f = *(const float4*)(x + (rowbase + r) * DIMD + c);
        Atile[r][c + 0] = (bf16_t)f.x;
        Atile[r][c + 1] = (bf16_t)f.y;
        Atile[r][c + 2] = (bf16_t)f.z;
        Atile[r][c + 3] = (bf16_t)f.w;
    }
    __syncthreads();

    const int wave = tid >> 5;
    const int lane = tid & 31;
    const int lrow = lane & 15;   // N column within tile / A row within M-tile
    const int hi   = lane >> 4;   // lane group

    float part[4][8];
#pragma unroll
    for (int mt = 0; mt < 4; ++mt)
#pragma unroll
        for (int i = 0; i < 8; ++i) part[mt][i] = 0.0f;

    for (int t = 0; t < 4; ++t) {
        const int hbase = (wave + 8 * t) * 16;       // 32 h-tiles over H=512
        v8f c[4] = {};
        const bf16_t* wrow = W1b + (size_t)(hbase + lrow) * DIMD + (hi ? 8 : 0);
#pragma unroll 2
        for (int k0 = 0; k0 < DIMD; k0 += 32) {
            // B 32x16 bf16: lane column n=lrow; lanes<16 K={k0..k0+7,k0+16..k0+23}; lanes>=16 +8
            ABfrag b;
            b.q[0] = *(const uint4*)(wrow + k0);
            b.q[1] = *(const uint4*)(wrow + k0 + 16);
#pragma unroll
            for (int mt = 0; mt < 4; ++mt) {
                // A 16x32 bf16: lanes<16 K={k0..k0+7,k0+16..k0+23}; lanes>=16 +8
                ABfrag a;
                const bf16_t* arow = &Atile[mt * 16 + lrow][k0 + hi * 8];
                a.q[0] = *(const uint4*)(arow);
                a.q[1] = *(const uint4*)(arow + 16);
                c[mt] = __builtin_amdgcn_wmma_f32_16x16x32_bf16(false, a.v, false, b.v,
                                                                (short)0, c[mt], false, false);
            }
        }
        // C layout: VGPR i, lane l -> M = mt*16 + i + 8*hi, N = lrow (h column)
        const float w2v = W2[hbase + lrow];
        const float bv  = b1[hbase + lrow];
#pragma unroll
        for (int mt = 0; mt < 4; ++mt)
#pragma unroll
            for (int i = 0; i < 8; ++i) {
                const float h = c[mt][i] + bv;
                const float g = 0.5f * h * (1.0f + erff(h * 0.70710678118654752f));
                part[mt][i] += g * w2v;
            }
    }

    // reduce over the 16 lanes sharing each M row
#pragma unroll
    for (int mt = 0; mt < 4; ++mt)
#pragma unroll
        for (int i = 0; i < 8; ++i) {
            float v = part[mt][i];
            v += __shfl_xor(v, 1, 32);
            v += __shfl_xor(v, 2, 32);
            v += __shfl_xor(v, 4, 32);
            v += __shfl_xor(v, 8, 32);
            part[mt][i] = v;
        }
    if (lrow == 0) {  // lanes 0 (i->M=mt*16+i) and 16 (i->M=mt*16+8+i)
#pragma unroll
        for (int mt = 0; mt < 4; ++mt)
#pragma unroll
            for (int i = 0; i < 8; ++i)
                spart[wave][mt * 16 + hi * 8 + i] = part[mt][i];
    }
    __syncthreads();
    if (tid < 64) {
        float s = b2[0];
#pragma unroll
        for (int w = 0; w < 8; ++w) s += spart[w][tid];
        scores[rowbase + tid] = s;
    }
}

// ---------------------------------------------------------------------------
// Top-200 per batch via single in-LDS bitonic sort of packed 64-bit keys.
// key = (~sortable(score) << 32) | idx  => ascending sort == descending score,
// ties -> lowest index (matches jax.lax.top_k / torch.topk order).
// ---------------------------------------------------------------------------
__global__ __launch_bounds__(256) void topk_kernel(const float* __restrict__ scores,
                                                   int* __restrict__ topk) {
    __shared__ unsigned long long keys[DIMN];   // 32 KB
    const int b = blockIdx.x, tid = threadIdx.x;
    for (int i = tid; i < DIMN; i += 256) {
        unsigned u = __float_as_uint(scores[(size_t)b * DIMN + i]);
        u = (u & 0x80000000u) ? ~u : (u | 0x80000000u);   // ascending-sortable
        keys[i] = ((unsigned long long)(~u) << 32) | (unsigned)i;
    }
    __syncthreads();
    for (int k = 2; k <= DIMN; k <<= 1) {
        for (int j = k >> 1; j > 0; j >>= 1) {
            for (int i = tid; i < DIMN; i += 256) {
                const int ixj = i ^ j;
                if (ixj > i) {
                    const unsigned long long a = keys[i], c = keys[ixj];
                    const bool up = ((i & k) == 0);
                    if ((a > c) == up) { keys[i] = c; keys[ixj] = a; }
                }
            }
            __syncthreads();
        }
    }
    for (int i = tid; i < KSEL; i += 256)
        topk[b * KSEL + i] = (int)(keys[i] & 0xFFFFFFFFu);
}

// ---------------------------------------------------------------------------
// Gather selected tokens, convert to bf16. One block per selected row.
// ---------------------------------------------------------------------------
__global__ __launch_bounds__(256) void gather_kernel(const float* __restrict__ x,
                                                     const int* __restrict__ topk,
                                                     bf16_t* __restrict__ xg) {
    const int row = blockIdx.x;              // 0..3199 = b*200+k
    const int b   = row / KSEL;
    const int idx = topk[row];
    const float* src = x + ((size_t)b * DIMN + idx) * DIMD;
    bf16_t* dst = xg + (size_t)row * DIMD;
    for (int i = threadIdx.x * 4; i < DIMD; i += 256 * 4) {
        const float4 f = *(const float4*)(src + i);
        dst[i + 0] = (bf16_t)f.x;
        dst[i + 1] = (bf16_t)f.y;
        dst[i + 2] = (bf16_t)f.z;
        dst[i + 3] = (bf16_t)f.w;
    }
}

// ---------------------------------------------------------------------------
// Final projection: out[3200,4096] = xg[3200,1024] . Wp^T + bp   (fp32 out)
// grid = (50 m-blocks of 64 rows, 8 n-groups of 512 cols).
// A tile staged with CDNA5 async global->LDS copies (bf16 already in ws).
// ---------------------------------------------------------------------------
__global__ __launch_bounds__(256) void proj_kernel(const bf16_t* __restrict__ xg,
                                                   const bf16_t* __restrict__ Wpb,
                                                   const float* __restrict__ bp,
                                                   float* __restrict__ out) {
    __shared__ bf16_t Atile[64][DIMD + 8];   // 129 KB
    const int  mblk    = blockIdx.x;               // 0..49
    const int  ngrp    = blockIdx.y;               // 0..7
    const long rowbase = (long)mblk * 64;
    const int  tid     = threadIdx.x;

    // stage 64 x 1024 bf16 A tile via async global->LDS (16B per op)
    for (int i = tid; i < 64 * DIMD / 8; i += 256) {
        const int r  = (i * 8) / DIMD;
        const int cc = (i * 8) % DIMD;
        const unsigned lds_addr = (unsigned)(uintptr_t)&Atile[r][cc];
        const unsigned long long gaddr =
            (unsigned long long)(uintptr_t)(xg + (rowbase + r) * DIMD + cc);
        asm volatile("global_load_async_to_lds_b128 %0, %1, off"
                     :: "v"(lds_addr), "v"(gaddr) : "memory");
    }
    asm volatile("s_wait_asynccnt 0x0" ::: "memory");
    __syncthreads();

    const int wave = tid >> 5;
    const int lane = tid & 31;
    const int lrow = lane & 15;
    const int hi   = lane >> 4;

    for (int t = 0; t < 4; ++t) {
        const int nbase = (ngrp * 32 + wave * 4 + t) * 16;   // output column base
        v8f c[4] = {};
        const bf16_t* wrow = Wpb + (size_t)(nbase + lrow) * DIMD + (hi ? 8 : 0);
#pragma unroll 2
        for (int k0 = 0; k0 < DIMD; k0 += 32) {
            ABfrag b;
            b.q[0] = *(const uint4*)(wrow + k0);
            b.q[1] = *(const uint4*)(wrow + k0 + 16);
#pragma unroll
            for (int mt = 0; mt < 4; ++mt) {
                ABfrag a;
                const bf16_t* arow = &Atile[mt * 16 + lrow][k0 + hi * 8];
                a.q[0] = *(const uint4*)(arow);
                a.q[1] = *(const uint4*)(arow + 16);
                c[mt] = __builtin_amdgcn_wmma_f32_16x16x32_bf16(false, a.v, false, b.v,
                                                                (short)0, c[mt], false, false);
            }
        }
        const float bias = bp[nbase + lrow];
#pragma unroll
        for (int mt = 0; mt < 4; ++mt)
#pragma unroll
            for (int i = 0; i < 8; ++i) {
                out[(rowbase + mt * 16 + i + 8 * hi) * (long)DIMO + nbase + lrow] =
                    c[mt][i] + bias;
            }
    }
}

// ---------------------------------------------------------------------------
extern "C" void kernel_launch(void* const* d_in, const int* in_sizes, int n_in,
                              void* d_out, int out_size, void* d_ws, size_t ws_size,
                              hipStream_t stream) {
    const float* x  = (const float*)d_in[0];   // [16,4096,1024]
    const float* W1 = (const float*)d_in[1];   // [512,1024]
    const float* b1 = (const float*)d_in[2];   // [512]
    const float* W2 = (const float*)d_in[3];   // [1,512]
    const float* b2 = (const float*)d_in[4];   // [1]
    const float* Wp = (const float*)d_in[5];   // [4096,1024]
    const float* bp = (const float*)d_in[6];   // [4096]
    float* out = (float*)d_out;                // [16,200,4096]

    char* ws = (char*)d_ws;
    bf16_t* W1b    = (bf16_t*)(ws);                          // 1 MB
    bf16_t* Wpb    = (bf16_t*)(ws + (1u << 20));             // 8 MB
    float*  scores = (float*) (ws + (9u << 20));             // 256 KB
    int*    topk   = (int*)   (ws + (9u << 20) + (1u << 18));// 12.8 KB
    bf16_t* xg     = (bf16_t*)(ws + (10u << 20));            // 6.4 MB

    cvt_bf16_kernel<<<512,  256, 0, stream>>>(W1, W1b, DIMH * DIMD);
    cvt_bf16_kernel<<<2048, 256, 0, stream>>>(Wp, Wpb, DIMO * DIMD);

    score_kernel<<<DIMB * DIMN / 64, 256, 0, stream>>>(x, W1b, b1, W2, b2, scores);

    topk_kernel<<<DIMB, 256, 0, stream>>>(scores, topk);

    gather_kernel<<<DIMB * KSEL, 256, 0, stream>>>(x, topk, xg);

    proj_kernel<<<dim3(DIMB * KSEL / 64, 8), 256, 0, stream>>>(xg, Wpb, bp, out);
}
